// EncoderDecoderTransformer_51410758533144
// MI455X (gfx1250) — matrix operand order repaired and checked by reading, not compiled, not run on hardware
//
#include <hip/hip_runtime.h>

typedef __attribute__((ext_vector_type(16))) _Float16 v16h;
typedef __attribute__((ext_vector_type(8)))  _Float16 v8h;
typedef __attribute__((ext_vector_type(4)))  _Float16 v4h;
typedef __attribute__((ext_vector_type(8)))  float    v8f;

namespace {
constexpr int kB = 4, kS = 512, kD = 1024, kH = 16, kDK = 64, kDH = 4096, kNB = 4, kV = 32000;
constexpr int kBS = kB * kS;   // 2048 token rows
constexpr int kBH = kB * kH;   // 64 (batch,head) pairs
constexpr int TM = 128, TN = 128, TK = 32;
constexpr int LDSS = TK + 8;   // LDS row stride in halfs (80B -> 16B-aligned fragments)
}

// ---------------------------------------------------------------------------
// CDNA5 async global->LDS path (ASYNCcnt), with synchronous fallback.
// (probe-verified: param 1 is 'int __vector(4) __device__*', generic pointers
// convert implicitly via CUDA LangAS rules)
// ---------------------------------------------------------------------------
#if defined(__has_builtin)
#if __has_builtin(__builtin_amdgcn_global_load_async_to_lds_b128)
#define HAVE_ASYNC_LDS 1
#endif
#endif
#ifndef HAVE_ASYNC_LDS
#define HAVE_ASYNC_LDS 0
#endif

#if HAVE_ASYNC_LDS
typedef int async_v4i __attribute__((vector_size(16)));
__device__ __forceinline__ void async_cp16(const void* g, void* l) {
  __builtin_amdgcn_global_load_async_to_lds_b128(
      (async_v4i*)g, (async_v4i*)l, 0, 0);
}
#if __has_builtin(__builtin_amdgcn_s_wait_asynccnt)
#define WAIT_ASYNC0() __builtin_amdgcn_s_wait_asynccnt(0)
#else
#define WAIT_ASYNC0() asm volatile("s_wait_asynccnt 0x0" ::: "memory")
#endif
#endif

// ---------------------------------------------------------------------------
// CDNA5 Tensor Data Mover (TENSORcnt): one descriptor moves a whole B tile
// (128 rows x 64B) into LDS, with D# pad fields reproducing the LDSS-padded
// layout (pad_interval: 16 DWORDs of data -> pad_amount: 4 DWORDs).
// ---------------------------------------------------------------------------
#if defined(__has_builtin)
#if __has_builtin(__builtin_amdgcn_tensor_load_to_lds)
#define HAVE_TDM 1
#endif
#endif
#ifndef HAVE_TDM
#define HAVE_TDM 0
#endif

__device__ __forceinline__ void tdm_issue_b_tile(const void* gsrc, void* ldst,
                                                 int tensor_k, int tensor_n, long stride_elems)
{
#if HAVE_TDM
  typedef int tdm_v4i __attribute__((vector_size(16)));
  typedef int tdm_v8i __attribute__((vector_size(32)));
  const unsigned long long ga = (unsigned long long)(uintptr_t)gsrc;
  const unsigned int       la = (unsigned int)(uintptr_t)ldst;   // flat LDS addr[31:0]
  const long long          st0 = (long long)stride_elems;
  // D# group 0: count=1 | lds_addr | global_addr[56:0] | type=2
  tdm_v4i g0 = { 1,
                 (int)la,
                 (int)(unsigned int)ga,
                 (int)((unsigned int)((ga >> 32) & 0x1FFFFFFu) | (2u << 30)) };
  // D# group 1: data_size=2B(1), pad_enable, pad_interval=16DW(3), pad_amount=4DW(3),
  //             tensor_dim0/1, tile_dim0=32, tile_dim1=128, tensor_dim0_stride
  tdm_v8i g1 = { (int)((1u << 16) | (1u << 20) | (3u << 22) | (3u << 25)),
                 (int)(((unsigned)tensor_k & 0xFFFFu) << 16),
                 (int)((((unsigned)tensor_k >> 16) & 0xFFFFu) | (((unsigned)tensor_n & 0xFFFFu) << 16)),
                 (int)((((unsigned)tensor_n >> 16) & 0xFFFFu) | (32u << 16)),
                 (int)128,
                 (int)(unsigned int)(st0 & 0xFFFFFFFFll),
                 (int)(unsigned int)((st0 >> 32) & 0xFFFFll),
                 0 };
  tdm_v4i z4 = {0, 0, 0, 0};
#if __clang_major__ >= 23
  tdm_v8i z8 = {0, 0, 0, 0, 0, 0, 0, 0};
  __builtin_amdgcn_tensor_load_to_lds(g0, g1, z4, z4, z8, 0);
#else
  __builtin_amdgcn_tensor_load_to_lds(g0, g1, z4, z4, 0);
#endif
#else
  (void)gsrc; (void)ldst; (void)tensor_k; (void)tensor_n; (void)stride_elems;
#endif
}

__device__ __forceinline__ void tdm_wait0()
{
#if HAVE_TDM
#if __has_builtin(__builtin_amdgcn_s_wait_tensorcnt)
  __builtin_amdgcn_s_wait_tensorcnt(0);
#else
  asm volatile("s_wait_tensorcnt 0x0" ::: "memory");
#endif
#endif
}

// ---------------------------------------------------------------------------
// One-time per-call weight prep: W[K,N] f32 -> W^T[N,K] f16 (LDS tile transpose)
// ---------------------------------------------------------------------------
__global__ __launch_bounds__(256) void transpose_cvt_kernel(
    const float* __restrict__ in, _Float16* __restrict__ out, int K, int N)
{
  __shared__ float tile[32][33];
  const int n0 = blockIdx.x * 32, k0 = blockIdx.y * 32;
  const int tx = threadIdx.x & 31, ty = threadIdx.x >> 5;   // 32 x 8
#pragma unroll
  for (int i = 0; i < 4; ++i) {
    int k = ty + i * 8;
    tile[k][tx] = in[(size_t)(k0 + k) * N + (n0 + tx)];
  }
  __syncthreads();
#pragma unroll
  for (int i = 0; i < 4; ++i) {
    int n = ty + i * 8;
    out[(size_t)(n0 + n) * K + (k0 + tx)] = (_Float16)tile[tx][n];
  }
}

// ---------------------------------------------------------------------------
// WMMA GEMM, double-buffered LDS, software-pipelined.
//   C[M,N] = act( A[M,K](f32) * B + bias ),  f32 accumulate via
//   V_WMMA_F32_16X16X32_F16.
// BMODE 0: B = f32 [K,N] row-major (transpose-staged); GUARDed (P@V, N=64)
// BMODE 2: B = f16 [N,K] row-major, per-lane async-to-LDS staging (QK^T)
// BMODE 3: B = f16 [N,K] row-major, Tensor Data Mover staging (weight GEMMs)
// Optional C16: also emit f16 copy of C (feeds later f16-B GEMMs).
// 256 threads = 8 wave32 (2x4); each wave: 64x32 = 4x2 WMMA tiles.
// ---------------------------------------------------------------------------
template <int BMODE, bool GUARD>
__global__ __launch_bounds__(256) void gemm_wmma_kernel(
    const float* __restrict__ A, long lda, long strideA,
    const void* __restrict__ Bv, long ldb, long strideB,
    float* __restrict__ C, long ldc, long strideC,
    _Float16* __restrict__ C16,
    int M, int N, int K,
    const float* __restrict__ bias, int relu)
{
  __shared__ __align__(16) _Float16 As[2][TM][LDSS];
  __shared__ __align__(16) _Float16 Bs[2][TN][LDSS];   // Bs[n][k]

  constexpr bool USE_TDM = (BMODE == 3) && (HAVE_TDM != 0);

  const int tid = threadIdx.x;
  const int bz  = blockIdx.z;
  A += (size_t)bz * strideA;
  const float*    Bf = (const float*)Bv    + (size_t)bz * strideB;
  const _Float16* Bh = (const _Float16*)Bv + (size_t)bz * strideB;
  C += (size_t)bz * strideC;
  if (C16) C16 += (size_t)bz * strideC;

  const int m0 = blockIdx.y * TM;
  const int n0 = blockIdx.x * TN;
  const int wave = tid >> 5, lane = tid & 31;
  const int wm = (wave >> 2) * 64, wn = (wave & 3) * 32;
  const int hi16 = lane >> 4, lr = lane & 15;

  float4 areg[4];
  float4 breg[4];     // BMODE 0
  uint4  hreg[2];     // BMODE 2/3 sync fallback
  (void)breg; (void)hreg;

  auto issue_tile = [&](int k0, int buf) {
    (void)buf;
#pragma unroll
    for (int it = 0; it < 4; ++it) {         // A: 128x32 f32, 4 float4/thread
      int idx = tid + it * 256;
      int row = idx >> 3, kq = (idx & 7) << 2;
      areg[it] = *reinterpret_cast<const float4*>(A + (size_t)(m0 + row) * lda + (k0 + kq));
    }
    if constexpr (USE_TDM) {
      if (wave == 0)                          // one TDM descriptor per tile
        tdm_issue_b_tile(Bh + (size_t)n0 * ldb + k0, &Bs[buf][0][0], K, N, ldb);
    } else if constexpr (BMODE == 0) {
#pragma unroll
      for (int it = 0; it < 4; ++it) {       // B: [K,N] rows, transpose at stage
        int idx = tid + it * 256;
        int kk = idx >> 5, nq = (idx & 31) << 2;
        const float* p = Bf + (size_t)(k0 + kk) * ldb + (n0 + nq);
        if (!GUARD || (n0 + nq + 3 < N)) {
          breg[it] = *reinterpret_cast<const float4*>(p);
        } else {
          breg[it] = make_float4((n0 + nq + 0 < N) ? p[0] : 0.f,
                                 (n0 + nq + 1 < N) ? p[1] : 0.f,
                                 (n0 + nq + 2 < N) ? p[2] : 0.f,
                                 (n0 + nq + 3 < N) ? p[3] : 0.f);
        }
      }
    } else {
#pragma unroll
      for (int it = 0; it < 2; ++it) {       // B: f16 [N,K] rows, 16B segments
        int idx = tid + it * 256;
        int n = idx >> 2, sg = (idx & 3) << 3;
        const _Float16* g = Bh + (size_t)(n0 + n) * ldb + (k0 + sg);
#if HAVE_ASYNC_LDS
        async_cp16(g, &Bs[buf][n][sg]);      // per-lane DMA into LDS
#else
        hreg[it] = *reinterpret_cast<const uint4*>(g);
#endif
      }
    }
  };

  auto stage_tile = [&](int buf) {
#pragma unroll
    for (int it = 0; it < 4; ++it) {         // cvt A regs -> f16 LDS
      int idx = tid + it * 256;
      int row = idx >> 3, kq = (idx & 7) << 2;
      v4h h = {(_Float16)areg[it].x, (_Float16)areg[it].y,
               (_Float16)areg[it].z, (_Float16)areg[it].w};
      *reinterpret_cast<v4h*>(&As[buf][row][kq]) = h;
    }
    if constexpr (USE_TDM) {
      if (wave == 0) tdm_wait0();            // tile landed in LDS
    } else if constexpr (BMODE == 0) {
#pragma unroll
      for (int it = 0; it < 4; ++it) {
        int idx = tid + it * 256;
        int kk = idx >> 5, nq = (idx & 31) << 2;
        Bs[buf][nq + 0][kk] = (_Float16)breg[it].x;
        Bs[buf][nq + 1][kk] = (_Float16)breg[it].y;
        Bs[buf][nq + 2][kk] = (_Float16)breg[it].z;
        Bs[buf][nq + 3][kk] = (_Float16)breg[it].w;
      }
    } else {
#if HAVE_ASYNC_LDS
      WAIT_ASYNC0();                          // copies for this buffer complete
#else
#pragma unroll
      for (int it = 0; it < 2; ++it) {
        int idx = tid + it * 256;
        int n = idx >> 2, sg = (idx & 3) << 3;
        *reinterpret_cast<uint4*>(&Bs[buf][n][sg]) = hreg[it];
      }
#endif
    }
  };

  v8f acc[4][2] = {};
  const int nt = K / TK;                      // all K are multiples of 32

  issue_tile(0, 0);
  for (int t = 0; t < nt; ++t) {
    const int cur = t & 1;
    stage_tile(cur);
    __syncthreads();
    if (t + 1 < nt) issue_tile((t + 1) * TK, cur ^ 1);

    // fragments per ISA 16-bit A/B layouts, contiguous ds_load_b128s
    v16h afrag[4], bfrag[2];
#pragma unroll
    for (int mf = 0; mf < 4; ++mf) {
      const _Float16* rp = &As[cur][wm + mf * 16 + lr][0];
      v8h lo = *reinterpret_cast<const v8h*>(rp + hi16 * 8);
      v8h hi = *reinterpret_cast<const v8h*>(rp + 16 + hi16 * 8);
      afrag[mf] = __builtin_shufflevector(lo, hi, 0, 1, 2, 3, 4, 5, 6, 7,
                                          8, 9, 10, 11, 12, 13, 14, 15);
    }
#pragma unroll
    for (int nf = 0; nf < 2; ++nf) {
      const _Float16* rp = &Bs[cur][wn + nf * 16 + lr][0];
      v8h lo = *reinterpret_cast<const v8h*>(rp + hi16 * 8);
      v8h hi = *reinterpret_cast<const v8h*>(rp + 16 + hi16 * 8);
      bfrag[nf] = __builtin_shufflevector(lo, hi, 0, 1, 2, 3, 4, 5, 6, 7,
                                          8, 9, 10, 11, 12, 13, 14, 15);
    }
#pragma unroll
    for (int mf = 0; mf < 4; ++mf)
#pragma unroll
      for (int nf = 0; nf < 2; ++nf)
        acc[mf][nf] = __builtin_amdgcn_wmma_f32_16x16x32_f16(
            false, afrag[mf], false, bfrag[nf], (short)0, acc[mf][nf], false, false);
  }

  // epilogue: bias / relu / optional f16 mirror
#pragma unroll
  for (int mf = 0; mf < 4; ++mf) {
#pragma unroll
    for (int nf = 0; nf < 2; ++nf) {
      int col = n0 + wn + nf * 16 + lr;
      if (GUARD && col >= N) continue;
      float bv = bias ? bias[col] : 0.f;
      int rbase = m0 + wm + mf * 16 + hi16 * 8;
#pragma unroll
      for (int r = 0; r < 8; ++r) {
        int gr = rbase + r;
        float v = acc[mf][nf][r] + bv;
        if (relu) v = fmaxf(v, 0.f);
        C[(size_t)gr * ldc + col] = v;
        if (C16) C16[(size_t)gr * ldc + col] = (_Float16)v;
      }
    }
  }
}

// ---------------------------------------------------------------------------
// token embed * sqrt(D) + sinusoidal positional encoding
// ---------------------------------------------------------------------------
__global__ __launch_bounds__(512) void embed_kernel(
    const int* __restrict__ tok, const float* __restrict__ emb,
    float* __restrict__ out)
{
  const int bs = blockIdx.x;
  const int s  = bs % kS;
  const int t  = threadIdx.x;                 // pair 0..511
  const long tb = (long)tok[bs] * kD;
  const long ob = (long)bs * kD;
  const float div = __expf(-(float)(2 * t) * (9.2103403719761836f / (float)kD));
  const float ang = (float)s * div;
  out[ob + 2 * t]     = emb[tb + 2 * t]     * 32.f + __sinf(ang);
  out[ob + 2 * t + 1] = emb[tb + 2 * t + 1] * 32.f + __cosf(ang);
}

// ---------------------------------------------------------------------------
// row softmax over 512 cols, fused 1/sqrt(D) scale + causal mask
// ---------------------------------------------------------------------------
__global__ __launch_bounds__(256) void softmax_kernel(
    float* __restrict__ t, int rows_per_mat, int causal, float scale)
{
  const int tid = threadIdx.x;
  const long row = blockIdx.x;
  const int r = (int)(row % rows_per_mat);
  float* p = t + row * kS;
  __shared__ float red[256];
  int c0 = tid, c1 = tid + 256;
  float v0 = p[c0] * scale; if (causal && c0 > r) v0 -= 1e9f;
  float v1 = p[c1] * scale; if (causal && c1 > r) v1 -= 1e9f;
  red[tid] = fmaxf(v0, v1); __syncthreads();
  for (int s = 128; s > 0; s >>= 1) {
    if (tid < s) red[tid] = fmaxf(red[tid], red[tid + s]);
    __syncthreads();
  }
  float mx = red[0]; __syncthreads();
  v0 = __expf(v0 - mx); v1 = __expf(v1 - mx);
  red[tid] = v0 + v1; __syncthreads();
  for (int s = 128; s > 0; s >>= 1) {
    if (tid < s) red[tid] += red[tid + s];
    __syncthreads();
  }
  float inv = 1.f / red[0];
  p[c0] = v0 * inv;
  p[c1] = v1 * inv;
}

// ---------------------------------------------------------------------------
// out = LayerNorm(a + res) * g + b   (D=1024, one block per row; in-place ok)
// ---------------------------------------------------------------------------
__global__ __launch_bounds__(256) void add_layernorm_kernel(
    const float* __restrict__ a, const float* __restrict__ res,
    const float* __restrict__ g, const float* __restrict__ b,
    float* __restrict__ out)
{
  const int tid = threadIdx.x;
  const long base = (long)blockIdx.x * kD;
  __shared__ float red[256];
  float x[4];
  float s = 0.f;
#pragma unroll
  for (int i = 0; i < 4; ++i) {
    int c = tid + i * 256;
    x[i] = a[base + c] + res[base + c];
    s += x[i];
  }
  red[tid] = s; __syncthreads();
  for (int st = 128; st > 0; st >>= 1) {
    if (tid < st) red[tid] += red[tid + st];
    __syncthreads();
  }
  float mu = red[0] * (1.f / (float)kD); __syncthreads();
  float vs = 0.f;
#pragma unroll
  for (int i = 0; i < 4; ++i) { float d = x[i] - mu; vs += d * d; }
  red[tid] = vs; __syncthreads();
  for (int st = 128; st > 0; st >>= 1) {
    if (tid < st) red[tid] += red[tid + st];
    __syncthreads();
  }
  float inv = rsqrtf(red[0] * (1.f / (float)kD) + 1e-5f);
#pragma unroll
  for (int i = 0; i < 4; ++i) {
    int c = tid + i * 256;
    out[base + c] = g[c] * (x[i] - mu) * inv + b[c];
  }
}

// ---------------------------------------------------------------------------
// host-side helpers
// ---------------------------------------------------------------------------
static void launch_gemm(hipStream_t st,
                        const float* A, long lda, long sA,
                        const void* B, long ldb, long sB,
                        float* C, long ldc, long sC, _Float16* C16,
                        int M, int N, int K, int batch,
                        const float* bias, bool relu, int bmode)
{
  dim3 grid((N + TN - 1) / TN, (M + TM - 1) / TM, batch);
  if (bmode == 3)
    gemm_wmma_kernel<3, false><<<grid, 256, 0, st>>>(A, lda, sA, B, ldb, sB,
                                                     C, ldc, sC, C16, M, N, K, bias, relu ? 1 : 0);
  else if (bmode == 2)
    gemm_wmma_kernel<2, false><<<grid, 256, 0, st>>>(A, lda, sA, B, ldb, sB,
                                                     C, ldc, sC, C16, M, N, K, bias, relu ? 1 : 0);
  else
    gemm_wmma_kernel<0, true><<<grid, 256, 0, st>>>(A, lda, sA, B, ldb, sB,
                                                    C, ldc, sC, C16, M, N, K, bias, relu ? 1 : 0);
}

static _Float16* conv_w(hipStream_t st, const float* w, int K, int N, _Float16** cur)
{
  _Float16* out = *cur; *cur += (size_t)K * N;
  transpose_cvt_kernel<<<dim3(N / 32, K / 32), 256, 0, st>>>(w, out, K, N);
  return out;
}

// Heads split by plain reshape -> head (b,h) is the contiguous [S,DK] block at
// (b*H+h)*S*DK of the flat projection buffer; batched GEMMs, stride S*DK.
static void run_mha(hipStream_t st, const float* qin, const float* kvin,
                    const _Float16* wqT, const _Float16* wkT,
                    const _Float16* wvT, const _Float16* woT,
                    float* qp, float* kp, float* vp, float* t3, float* out,
                    float* lgt, _Float16* kp16, int causal)
{
  launch_gemm(st, qin,  kD, 0, wqT, kD, 0, qp, kD, 0, nullptr, kBS, kD, kD, 1, nullptr, false, 3);
  launch_gemm(st, kvin, kD, 0, wkT, kD, 0, kp, kD, 0, kp16,   kBS, kD, kD, 1, nullptr, false, 3);
  launch_gemm(st, kvin, kD, 0, wvT, kD, 0, vp, kD, 0, nullptr, kBS, kD, kD, 1, nullptr, false, 3);
  const long hs = (long)kS * kDK;
  // logits = Q @ K^T : B = kp16 as f16 [N=S, K=DK] rows -> per-lane async path
  launch_gemm(st, qp, kDK, hs, kp16, kDK, hs, lgt, kS, (long)kS * kS, nullptr,
              kS, kS, kDK, kBH, nullptr, false, 2);
  softmax_kernel<<<dim3(kBH * kS), 256, 0, st>>>(lgt, kS, causal, 1.f / 32.f);
  // attn = P @ V : f32 [K,N] path, N=64 guarded
  launch_gemm(st, lgt, kS, (long)kS * kS, vp, kDK, hs, t3, kDK, hs, nullptr,
              kS, kDK, kS, kBH, nullptr, false, 0);
  launch_gemm(st, t3, kD, 0, woT, kD, 0, out, kD, 0, nullptr, kBS, kD, kD, 1, nullptr, false, 3);
}

extern "C" void kernel_launch(void* const* d_in, const int* in_sizes, int n_in,
                              void* d_out, int out_size, void* d_ws, size_t ws_size,
                              hipStream_t stream)
{
  (void)in_sizes; (void)n_in; (void)out_size; (void)ws_size;
  auto F = [&](int i) { return (const float*)d_in[i]; };

  // setup_inputs() insertion order:
  // 0 enc_input, 1 dec_input, 2 enc_pad, 3 dec_pad, 4 causal_mask,
  // 5 enc_emb, 6 dec_emb,
  // 7.. enc_blocks (12 each: wq wk wv wo w1 b1 w2 b2 ln1g ln1b ln2g ln2b)
  // 55.. dec_blocks (18 each: mmha wq..wo, mha wq..wo, w1 b1 w2 b2, ln1/2/3 g,b)
  // 127 proj_w, 128 proj_b
  const int* enc_tok = (const int*)d_in[0];
  const int* dec_tok = (const int*)d_in[1];

  float* ws = (float*)d_ws;
  float* x   = ws; ws += (size_t)kBS * kD;      // encoder activations / enc_out
  float* y   = ws; ws += (size_t)kBS * kD;      // decoder activations
  float* qp  = ws; ws += (size_t)kBS * kD;
  float* kp  = ws; ws += (size_t)kBS * kD;
  float* vp  = ws; ws += (size_t)kBS * kD;
  float* t3  = ws; ws += (size_t)kBS * kD;      // attention context
  float* t0  = ws; ws += (size_t)kBS * kD;      // MHA out / FF out
  float* ffb = ws; ws += (size_t)kBS * kDH;     // FF hidden
  float* lgt = ws; ws += (size_t)kBH * kS * kS; // attention logits/probs

  _Float16* hcur = (_Float16*)ws;
  _Float16* kp16 = hcur; hcur += (size_t)kBS * kD;

  // ---- one-time per call: all GEMM weights -> transposed f16 ----
  _Float16 *encw[kNB][6], *decw[kNB][10];
  for (int bl = 0; bl < kNB; ++bl) {
    const int p = 7 + bl * 12;
    for (int j = 0; j < 4; ++j) encw[bl][j] = conv_w(stream, F(p + j), kD, kD, &hcur);
    encw[bl][4] = conv_w(stream, F(p + 4), kD, kDH, &hcur);   // w1
    encw[bl][5] = conv_w(stream, F(p + 6), kDH, kD, &hcur);   // w2
  }
  for (int bl = 0; bl < kNB; ++bl) {
    const int p = 55 + bl * 18;
    for (int j = 0; j < 8; ++j) decw[bl][j] = conv_w(stream, F(p + j), kD, kD, &hcur);
    decw[bl][8] = conv_w(stream, F(p + 8),  kD, kDH, &hcur);  // w1
    decw[bl][9] = conv_w(stream, F(p + 10), kDH, kD, &hcur);  // w2
  }
  _Float16* projT = conv_w(stream, F(127), kD, kV, &hcur);

  // ---------------- encoder ----------------
  embed_kernel<<<dim3(kBS), 512, 0, stream>>>(enc_tok, F(5), x);
  for (int bl = 0; bl < kNB; ++bl) {
    const int p = 7 + bl * 12;
    run_mha(stream, x, x, encw[bl][0], encw[bl][1], encw[bl][2], encw[bl][3],
            qp, kp, vp, t3, t0, lgt, kp16, /*causal=*/0);
    add_layernorm_kernel<<<dim3(kBS), 256, 0, stream>>>(t0, x, F(p + 8), F(p + 9), x);
    launch_gemm(stream, x, kD, 0, encw[bl][4], kD, 0, ffb, kDH, 0, nullptr,
                kBS, kDH, kD, 1, F(p + 5), /*relu=*/true, 3);
    launch_gemm(stream, ffb, kDH, 0, encw[bl][5], kDH, 0, t0, kD, 0, nullptr,
                kBS, kD, kDH, 1, F(p + 7), false, 3);
    add_layernorm_kernel<<<dim3(kBS), 256, 0, stream>>>(t0, x, F(p + 10), F(p + 11), x);
  }

  // ---------------- decoder ----------------
  embed_kernel<<<dim3(kBS), 512, 0, stream>>>(dec_tok, F(6), y);
  for (int bl = 0; bl < kNB; ++bl) {
    const int p = 55 + bl * 18;
    run_mha(stream, y, y, decw[bl][0], decw[bl][1], decw[bl][2], decw[bl][3],
            qp, kp, vp, t3, t0, lgt, kp16, /*causal=*/1);
    add_layernorm_kernel<<<dim3(kBS), 256, 0, stream>>>(t0, y, F(p + 12), F(p + 13), y);
    run_mha(stream, y, x, decw[bl][4], decw[bl][5], decw[bl][6], decw[bl][7],
            qp, kp, vp, t3, t0, lgt, kp16, /*causal=*/0);
    add_layernorm_kernel<<<dim3(kBS), 256, 0, stream>>>(t0, y, F(p + 14), F(p + 15), y);
    launch_gemm(stream, y, kD, 0, decw[bl][8], kD, 0, ffb, kDH, 0, nullptr,
                kBS, kDH, kD, 1, F(p + 9), true, 3);
    launch_gemm(stream, ffb, kDH, 0, decw[bl][9], kDH, 0, t0, kD, 0, nullptr,
                kBS, kD, kDH, 1, F(p + 11), false, 3);
    add_layernorm_kernel<<<dim3(kBS), 256, 0, stream>>>(t0, y, F(p + 16), F(p + 17), y);
  }

  // ---------------- output projection ----------------
  launch_gemm(stream, y, kD, 0, projT, kD, 0, (float*)d_out, kV, 0, nullptr,
              kBS, kV, kD, 1, F(128), false, 3);
}
